// GATv2Layer_43568148251357
// MI455X (gfx1250) — matrix operand order repaired and checked by reading, not compiled, not run on hardware
//
#include <hip/hip_runtime.h>

#define IN_F 256
#define OUT_F 128
#define E_F 64
#define NH 8
#define HD 16
#define NEG_SLOPE 0.2f

#if defined(__has_builtin)
#if __has_builtin(__builtin_amdgcn_global_load_async_to_lds_b32) && \
    __has_builtin(__builtin_amdgcn_s_wait_asynccnt)
#define HAVE_ASYNC_LDS 1
#endif
#endif
#ifndef HAVE_ASYNC_LDS
#define HAVE_ASYNC_LDS 0
#endif

// int->pointer casts into numbered address spaces (generic->AS casts are
// rejected by clang; int->AS-pointer casts are legal). Builtin params are
// 'int*' in AS1 (global) / AS3 (LDS) per the round-2 diagnostic.
#define AS_GLOBAL_I32(p) ((__attribute__((address_space(1))) int*)(__UINTPTR_TYPE__)(p))
#define AS_LOCAL_I32(p)  ((__attribute__((address_space(3))) int*)(__UINTPTR_TYPE__)(p))

typedef __attribute__((ext_vector_type(16))) __bf16 v16bf;
typedef __attribute__((ext_vector_type(8))) float v8f;

union Frag16 { v16bf v; unsigned u[8]; };

__device__ __forceinline__ unsigned short f2bf(float f) {
  unsigned u = __float_as_uint(f);
  u += 0x7FFFu + ((u >> 16) & 1u);   // round-to-nearest-even
  return (unsigned short)(u >> 16);
}
// order-preserving float->uint encoding for atomicMax
__device__ __forceinline__ unsigned encf(float f) {
  unsigned u = __float_as_uint(f);
  return (u & 0x80000000u) ? ~u : (u | 0x80000000u);
}
__device__ __forceinline__ float decf(unsigned u) {
  return (u & 0x80000000u) ? __uint_as_float(u & 0x7FFFFFFFu) : __uint_as_float(~u);
}
// ISA 7.12.2: 16-bit A 16x32 — lane half 0: K 0-7,16-23 ; half 1: K 8-15,24-31
__device__ __forceinline__ int a_koff(int j, int half) {
  return ((j < 4) ? 0 : 16) + half * 8 + 2 * (j & 3);
}

// ---------------- K0: weights -> bf16, transposed (Wt[n][k]) ----------------
__global__ void k_convert_weights(const float* __restrict__ Wl, const float* __restrict__ Wr,
                                  const float* __restrict__ Wv, const float* __restrict__ Wo,
                                  unsigned short* __restrict__ wt, unsigned short* __restrict__ wot) {
  int idx = blockIdx.x * blockDim.x + threadIdx.x;
  const int nproj = 3 * OUT_F * IN_F;
  if (idx < nproj) {
    int mat = idx / (OUT_F * IN_F);
    int rem = idx % (OUT_F * IN_F);
    int n = rem / IN_F;
    int k = rem % IN_F;
    const float* W = (mat == 0) ? Wl : ((mat == 1) ? Wr : Wv);
    wt[idx] = f2bf(W[k * OUT_F + n]);
  } else {
    int r = idx - nproj;
    if (r < OUT_F * OUT_F) {
      int n = r / OUT_F, k = r % OUT_F;
      wot[r] = f2bf(Wo[k * OUT_F + n]);
    }
  }
}

// ---------------- K1: left/right/values = X @ {Wl,Wr,Wv} + bias (WMMA) ------
__global__ void __launch_bounds__(256) k_proj_gemm(
    const float* __restrict__ X, const unsigned short* __restrict__ wt,
    const float* __restrict__ bl, const float* __restrict__ br, const float* __restrict__ bv,
    float* __restrict__ left, float* __restrict__ right, float* __restrict__ values, int N) {
  __shared__ unsigned short ldsA[16 * IN_F];  // 8 KB, bf16 A tile shared by 8 waves
  int row0 = blockIdx.x * 16;
  for (int i = threadIdx.x; i < 16 * IN_F; i += 256) {
    int r = i >> 8, c = i & (IN_F - 1);
    int row = row0 + r;
    float v = (row < N) ? X[(size_t)row * IN_F + c] : 0.f;
    ldsA[i] = f2bf(v);
  }
  __syncthreads();

  int lane = threadIdx.x & 31;
  int half = lane >> 4;
  int n = lane & 15;
  int ct = threadIdx.x >> 5;  // each wave owns one 16-column tile
  int m = n;                  // A row handled by this lane

  v8f acc0 = {}, acc1 = {}, acc2 = {};
  for (int kk = 0; kk < IN_F; kk += 32) {
    Frag16 a;
#pragma unroll
    for (int j = 0; j < 8; ++j)
      a.u[j] = *(const unsigned*)&ldsA[m * IN_F + kk + a_koff(j, half)];
    // B 32x16: lane half 0 holds K 0-7,16-23; half 1 holds K 8-15,24-31; Wt row-major in k
    const unsigned short* wcol = wt + (size_t)(ct * 16 + n) * IN_F + kk;
    Frag16 b0, b1, b2;
#pragma unroll
    for (int j = 0; j < 8; ++j) {
      b0.u[j] = *(const unsigned*)&wcol[a_koff(j, half)];
      b1.u[j] = *(const unsigned*)&wcol[(size_t)OUT_F * IN_F + a_koff(j, half)];
      b2.u[j] = *(const unsigned*)&wcol[(size_t)2 * OUT_F * IN_F + a_koff(j, half)];
    }
    acc0 = __builtin_amdgcn_wmma_f32_16x16x32_bf16(false, a.v, false, b0.v, (short)0, acc0, false, false);
    acc1 = __builtin_amdgcn_wmma_f32_16x16x32_bf16(false, a.v, false, b1.v, (short)0, acc1, false, false);
    acc2 = __builtin_amdgcn_wmma_f32_16x16x32_bf16(false, a.v, false, b2.v, (short)0, acc2, false, false);
  }

  int col = ct * 16 + n;
  float bb0 = bl[col], bb1 = br[col], bb2 = bv[col];
#pragma unroll
  for (int r = 0; r < 8; ++r) {  // D: m = r + 8*half, n = lane&15
    int row = row0 + half * 8 + r;
    if (row < N) {
      left[(size_t)row * OUT_F + col]   = acc0[r] + bb0;
      right[(size_t)row * OUT_F + col]  = acc1[r] + bb1;
      values[(size_t)row * OUT_F + col] = acc2[r] + bb2;
    }
  }
}

// ---------------- K3: per-edge GATv2 scores + segment max -------------------
__global__ void __launch_bounds__(256) k_edge_scores(
    const int* __restrict__ ei, const float* __restrict__ ef,
    const float* __restrict__ left, const float* __restrict__ right,
    const float* __restrict__ We, const float* __restrict__ be,
    const float* __restrict__ attn,
    float* __restrict__ scores, unsigned* __restrict__ maxenc, int Ecnt) {
  __shared__ float efs[2][E_F];
  int slot = threadIdx.x >> 7;          // 2 edges per 256-thread block
  int e = blockIdx.x * 2 + slot;
  int tl = threadIdx.x & 127;           // (h,d) = (tl/16, tl%16)

  // keep the streaming edge-feature read ahead of the waves
  if (tl == 0 && (e + 2) < Ecnt)
    __builtin_prefetch(ef + (size_t)(e + 2) * E_F, 0, 1);

#if HAVE_ASYNC_LDS
  // gfx1250 async global->LDS copy (ASYNCcnt path): pure byte copy, no VGPR data
  if (e < Ecnt && tl < E_F) {
    __builtin_amdgcn_global_load_async_to_lds_b32(
        AS_GLOBAL_I32(ef + (size_t)e * E_F + tl),
        AS_LOCAL_I32(&efs[slot][tl]), 0, 0);
  }
  __builtin_amdgcn_s_wait_asynccnt(0);
#else
  if (e < Ecnt && tl < E_F) efs[slot][tl] = ef[(size_t)e * E_F + tl];
#endif
  __syncthreads();
  if (e >= Ecnt) return;

  int h = tl >> 4, d = tl & 15;
  int src = ei[e], tgt = ei[Ecnt + e];
  float x = left[(size_t)tgt * OUT_F + tl] + right[(size_t)src * OUT_F + tl];
  float act = (x > 0.f) ? x : NEG_SLOPE * x;
  float p = act * attn[h * HD + d];
#pragma unroll
  for (int j = 0; j < 4; ++j) {         // fold edge bias partials into reduction
    int k = d + 16 * j;
    p += efs[slot][k] * We[k * NH + h];
  }
#pragma unroll
  for (int off = 8; off >= 1; off >>= 1)
    p += __shfl_xor(p, off, 16);        // 16-lane head-group reduction (wave32)
  if (d == 0) {
    float s = p + be[h];
    scores[(size_t)e * NH + h] = s;
    atomicMax(&maxenc[(size_t)tgt * NH + h], encf(s));
  }
}

// ---------------- K4: exp + unnormalized weighted scatter -------------------
__global__ void __launch_bounds__(256) k_edge_scatter(
    const int* __restrict__ ei, const float* __restrict__ scores,
    const unsigned* __restrict__ maxenc, const float* __restrict__ values,
    float* __restrict__ sumbuf, float* __restrict__ agg, int Ecnt) {
  long long idx = (long long)blockIdx.x * 256 + threadIdx.x;
  int e = (int)(idx >> 7);
  if (e >= Ecnt) return;
  int tl = (int)(idx & 127);
  int h = tl >> 4, d = tl & 15;
  int src = ei[e], tgt = ei[Ecnt + e];
  float s = scores[(size_t)e * NH + h];
  float mx = decf(maxenc[(size_t)tgt * NH + h]);
  float p = __expf(s - mx);
  if (d == 0) atomicAdd(&sumbuf[(size_t)tgt * NH + h], p);
  atomicAdd(&agg[(size_t)tgt * OUT_F + tl], p * values[(size_t)src * OUT_F + tl]);
}

// ---------------- K5: out = (agg / sum) @ Wo + bo (WMMA) --------------------
__global__ void __launch_bounds__(256) k_out_gemm(
    const float* __restrict__ agg, const float* __restrict__ sumbuf,
    const unsigned short* __restrict__ wot, const float* __restrict__ bo,
    float* __restrict__ out, int N) {
  __shared__ unsigned short ldsA[16 * OUT_F];
  int row0 = blockIdx.x * 16;
  for (int i = threadIdx.x; i < 16 * OUT_F; i += 256) {
    int r = i >> 7, c = i & (OUT_F - 1);
    int row = row0 + r;
    float v = 0.f;
    if (row < N)
      v = agg[(size_t)row * OUT_F + c] / (sumbuf[(size_t)row * NH + (c >> 4)] + 1e-10f);
    ldsA[i] = f2bf(v);
  }
  __syncthreads();

  int lane = threadIdx.x & 31;
  int half = lane >> 4;
  int n = lane & 15;
  int ct = threadIdx.x >> 5;
  int m = n;
  v8f acc = {};
  for (int kk = 0; kk < OUT_F; kk += 32) {
    Frag16 a, b;
#pragma unroll
    for (int j = 0; j < 8; ++j)
      a.u[j] = *(const unsigned*)&ldsA[m * OUT_F + kk + a_koff(j, half)];
    const unsigned short* wcol = wot + (size_t)(ct * 16 + n) * OUT_F + kk;
#pragma unroll
    for (int j = 0; j < 8; ++j)
      b.u[j] = *(const unsigned*)&wcol[a_koff(j, half)];
    acc = __builtin_amdgcn_wmma_f32_16x16x32_bf16(false, a.v, false, b.v, (short)0, acc, false, false);
  }
  int col = ct * 16 + n;
  float bb = bo[col];
#pragma unroll
  for (int r = 0; r < 8; ++r) {
    int row = row0 + half * 8 + r;
    if (row < N) out[(size_t)row * OUT_F + col] = acc[r] + bb;
  }
}

extern "C" void kernel_launch(void* const* d_in, const int* in_sizes, int n_in,
                              void* d_out, int out_size, void* d_ws, size_t ws_size,
                              hipStream_t stream) {
  const float* X    = (const float*)d_in[0];
  const int*   ei   = (const int*)d_in[1];
  const float* ef   = (const float*)d_in[2];
  const float* Wl   = (const float*)d_in[3];
  const float* bl   = (const float*)d_in[4];
  const float* Wr   = (const float*)d_in[5];
  const float* br   = (const float*)d_in[6];
  const float* We   = (const float*)d_in[7];
  const float* be   = (const float*)d_in[8];
  const float* attn = (const float*)d_in[9];
  const float* Wv   = (const float*)d_in[10];
  const float* bv   = (const float*)d_in[11];
  const float* Wo   = (const float*)d_in[12];
  const float* bo   = (const float*)d_in[13];
  float* out = (float*)d_out;

  int N = in_sizes[0] / IN_F;
  int E = in_sizes[1] / 2;

  char* base = (char*)d_ws;
  size_t off = 0;
  auto take = [&](size_t bytes) -> char* {
    char* p = base + off;
    off = (off + bytes + 255) & ~(size_t)255;
    return p;
  };
  float*          left   = (float*)take((size_t)N * OUT_F * 4);
  float*          right  = (float*)take((size_t)N * OUT_F * 4);
  float*          values = (float*)take((size_t)N * OUT_F * 4);
  float*          agg    = (float*)take((size_t)N * OUT_F * 4);
  float*          sumbuf = (float*)take((size_t)N * NH * 4);
  unsigned*       maxenc = (unsigned*)take((size_t)N * NH * 4);
  float*          scores = (float*)take((size_t)E * NH * 4);
  unsigned short* wt     = (unsigned short*)take((size_t)3 * OUT_F * IN_F * 2);
  unsigned short* wot    = (unsigned short*)take((size_t)OUT_F * OUT_F * 2);
  (void)ws_size; (void)n_in; (void)out_size;

  // zero-init accumulators (0 == lower bound of the order-preserving max encoding)
  (void)hipMemsetAsync(agg, 0, (size_t)N * OUT_F * 4, stream);
  (void)hipMemsetAsync(sumbuf, 0, (size_t)N * NH * 4, stream);
  (void)hipMemsetAsync(maxenc, 0, (size_t)N * NH * 4, stream);

  int convTotal = 3 * OUT_F * IN_F + OUT_F * OUT_F;
  k_convert_weights<<<(convTotal + 255) / 256, 256, 0, stream>>>(Wl, Wr, Wv, Wo, wt, wot);

  int gm = (N + 15) / 16;
  k_proj_gemm<<<gm, 256, 0, stream>>>(X, wt, bl, br, bv, left, right, values, N);

  k_edge_scores<<<(E + 1) / 2, 256, 0, stream>>>(ei, ef, left, right, We, be, attn,
                                                 scores, maxenc, E);

  long long tot = (long long)E * OUT_F;
  k_edge_scatter<<<(unsigned)((tot + 255) / 256), 256, 0, stream>>>(ei, scores, maxenc,
                                                                    values, sumbuf, agg, E);

  k_out_gemm<<<gm, 256, 0, stream>>>(agg, sumbuf, wot, bo, out, N);
}
